// Model_39676907885431
// MI455X (gfx1250) — matrix operand order repaired and checked by reading, not compiled, not run on hardware
//
#include <hip/hip_runtime.h>
#include <stdint.h>

typedef __attribute__((ext_vector_type(16))) _Float16 v16h;
typedef __attribute__((ext_vector_type(8)))  _Float16 v8h;
typedef __attribute__((ext_vector_type(8)))  float    v8f;
typedef __attribute__((ext_vector_type(4)))  float    v4f;

#define B_DIM 2
#define H_DIM 16
#define S_DIM 2048
#define D_DIM 64
#define QTILE 128        // queries per workgroup (8 waves x 16)
#define KBLK  64         // keys per softmax update
#define NBLK  (S_DIM / KBLK)
#define NWAVES 8

__device__ __forceinline__ uint32_t hash_u32(uint32_t x) {
    x ^= x >> 16; x *= 0x7feb352du;
    x ^= x >> 15; x *= 0x846ca68bu;
    x ^= x >> 16;
    return x;
}

__device__ __forceinline__ v16h make_a16(const _Float16* p0, const _Float16* p1) {
    v8h lo = *(const v8h*)p0;
    v8h hi = *(const v8h*)p1;
    v16h r;
#pragma unroll
    for (int i = 0; i < 8; ++i) { r[i] = lo[i]; r[i + 8] = hi[i]; }
    return r;
}

__global__ __launch_bounds__(256)
void fa_wmma_kernel(const float* __restrict__ q_g,
                    const float* __restrict__ kv_g,
                    float* __restrict__ out_g) {
    // QP overlay: pre-loop it holds the scaled Q tile; inside the loop the same
    // per-wave 2KB region holds that wave's P tile (each wave reads only its own
    // Q rows, so the overwrite is private to the wave -> no extra barriers).
    __shared__ _Float16 QP[QTILE * D_DIM];           // 16 KB
    __shared__ _Float16 Ks[2][KBLK][D_DIM];          // 16 KB keys row-major (ping-pong)
    __shared__ _Float16 Vt[2][D_DIM][KBLK];          // 16 KB values transposed (ping-pong)

    const int tid  = threadIdx.x;
    const int wave = tid >> 5;
    const int lane = tid & 31;
    const int hi   = lane >> 4;
    const int ln   = lane & 15;

    const int bh  = blockIdx.y;
    const int wgq = blockIdx.x * QTILE;

    const float* qptr  = q_g  + ((size_t)bh * S_DIM + wgq) * D_DIM;
    const float* kvptr = kv_g + (size_t)bh * S_DIM * D_DIM;
    float*       optr  = out_g + ((size_t)bh * S_DIM + wgq) * D_DIM;

    // ---- stage Q tile, f32 -> f16 with (1/sqrt(D))*log2(e) folded in ----
    const float sQ = 0.125f * 1.44269504f;
#pragma unroll
    for (int i = tid * 4; i < QTILE * D_DIM; i += 256 * 4) {
        v4f v = *(const v4f*)(qptr + i);
#pragma unroll
        for (int j = 0; j < 4; ++j) QP[i + j] = (_Float16)(v[j] * sQ);
    }
    // ---- stage K/V block 0 ----
#pragma unroll
    for (int j = 0; j < 4; ++j) {
        const int e = tid * 4 + j * 1024;
        v4f v = *(const v4f*)(kvptr + e);
        const int kr = e >> 6, dc = e & 63;
#pragma unroll
        for (int t = 0; t < 4; ++t) {
            const _Float16 h = (_Float16)v[t];
            Ks[0][kr][dc + t] = h;
            Vt[0][dc + t][kr] = h;
        }
    }
    __syncthreads();

    // ---- Q as WMMA B fragments (Bᵀ layout: N=q=ln, K=d contiguous) ----
    const int qrow = wave * 16 + ln;
    v16h bQ0 = *(const v16h*)&QP[qrow * D_DIM +  0 + hi * 16];
    v16h bQ1 = *(const v16h*)&QP[qrow * D_DIM + 32 + hi * 16];

    // ---- per-lane softmax state: this lane owns query q = ln ----
    float m_run = -1.0e38f, l_run = 0.0f;
    v8f acc[4];                       // Oᵀ tiles: M=d (r+8*hi), N=q=ln
#pragma unroll
    for (int n = 0; n < 4; ++n) acc[n] = v8f{};

    const int qg = wgq + wave * 16 + ln;
    const uint32_t rowbase = ((uint32_t)bh * (uint32_t)S_DIM + (uint32_t)qg) * (uint32_t)S_DIM;
    _Float16* Pmine = &QP[wave * (16 * D_DIM) + ln * D_DIM];   // this lane's P row (64 keys)

    for (int i = 0; i < NBLK; ++i) {
        const int cur = i & 1, nxt = cur ^ 1;
        const int kb  = i * KBLK;

        // ---- prefetch next K/V block into registers (latency hidden by WMMAs) ----
        v4f g[4];
        const bool have_next = (i + 1) < NBLK;
        if (have_next) {
#pragma unroll
            for (int j = 0; j < 4; ++j)
                g[j] = *(const v4f*)(kvptr + (size_t)(kb + KBLK) * D_DIM + tid * 4 + j * 1024);
        }

        // ---- Sᵀ = K·Qᵀ : 4 subtiles of 16 keys, K-dim = D = 64 ----
        v8f st[4];
#pragma unroll
        for (int s = 0; s < 4; ++s) st[s] = v8f{};
        {
            v16h aK[4][2];
#pragma unroll
            for (int s = 0; s < 4; ++s) {
                const int krow = s * 16 + ln;
                aK[s][0] = make_a16(&Ks[cur][krow][ 0 + hi * 16], &Ks[cur][krow][ 8 + hi * 16]);
                aK[s][1] = make_a16(&Ks[cur][krow][32 + hi * 16], &Ks[cur][krow][40 + hi * 16]);
            }
            st[0] = __builtin_amdgcn_wmma_f32_16x16x32_f16(false, aK[0][0], false, bQ0, (short)0, st[0], false, false);
            st[1] = __builtin_amdgcn_wmma_f32_16x16x32_f16(false, aK[1][0], false, bQ0, (short)0, st[1], false, true);
            st[2] = __builtin_amdgcn_wmma_f32_16x16x32_f16(false, aK[2][0], false, bQ0, (short)0, st[2], false, true);
            st[3] = __builtin_amdgcn_wmma_f32_16x16x32_f16(false, aK[3][0], false, bQ0, (short)0, st[3], false, true);
            st[0] = __builtin_amdgcn_wmma_f32_16x16x32_f16(false, aK[0][1], false, bQ1, (short)0, st[0], false, false);
            st[1] = __builtin_amdgcn_wmma_f32_16x16x32_f16(false, aK[1][1], false, bQ1, (short)0, st[1], false, true);
            st[2] = __builtin_amdgcn_wmma_f32_16x16x32_f16(false, aK[2][1], false, bQ1, (short)0, st[2], false, true);
            st[3] = __builtin_amdgcn_wmma_f32_16x16x32_f16(false, aK[3][1], false, bQ1, (short)0, st[3], false, true);
        }

        // ---- online softmax (exp2 domain), fused dropout + pack per subtile ----
        float mb = st[0][0];
#pragma unroll
        for (int s = 0; s < 4; ++s)
#pragma unroll
            for (int r = 0; r < 8; ++r) mb = fmaxf(mb, st[s][r]);
        mb = fmaxf(mb, __shfl_xor(mb, 16, 32));          // combine key halves
        const float mnew = fmaxf(m_run, mb);
        const float corr = __builtin_amdgcn_exp2f(m_run - mnew);
        m_run = mnew;

        float ps = 0.0f;
#pragma unroll
        for (int s = 0; s < 4; ++s) {
            float p[8];
#pragma unroll
            for (int r = 0; r < 8; ++r) {
                p[r] = __builtin_amdgcn_exp2f(st[s][r] - mnew);
                ps += p[r];
            }
            // dropout mask (deterministic hash stand-in for JAX PRNG);
            // 1/0.9 rescale folded into the epilogue divide.
            const uint32_t h0 = hash_u32(rowbase + (uint32_t)(kb + s * 16 + hi * 8));
            const uint32_t h1 = h0 * 0x9E3779B9u + 0x7F4A7C15u;
            v8h pk;
#pragma unroll
            for (int r = 0; r < 8; ++r) {
                const uint32_t byte = ((r < 4 ? h0 : h1) >> ((r & 3) * 8)) & 0xFFu;
                pk[r] = (byte < 230u) ? (_Float16)p[r] : (_Float16)0.0f;
            }
            *(v8h*)(Pmine + s * 16 + hi * 8) = pk;       // same-wave DS, in-order
        }
        ps += __shfl_xor(ps, 16, 32);
        l_run = l_run * corr + ps;                       // full softmax denominator
#pragma unroll
        for (int n = 0; n < 4; ++n)
#pragma unroll
            for (int r = 0; r < 8; ++r) acc[n][r] *= corr;

        // ---- Oᵀ += Vᵀ(16x32) · Pᵀ(32x16): 2 key passes x 4 d-tiles ----
#pragma unroll
        for (int p2 = 0; p2 < 2; ++p2) {
            v16h bp = make_a16(Pmine + p2 * 32 + hi * 8, Pmine + p2 * 32 + 16 + hi * 8);
            v16h av0 = make_a16(&Vt[cur][ 0 + ln][p2 * 32 + hi * 8], &Vt[cur][ 0 + ln][p2 * 32 + 16 + hi * 8]);
            v16h av1 = make_a16(&Vt[cur][16 + ln][p2 * 32 + hi * 8], &Vt[cur][16 + ln][p2 * 32 + 16 + hi * 8]);
            v16h av2 = make_a16(&Vt[cur][32 + ln][p2 * 32 + hi * 8], &Vt[cur][32 + ln][p2 * 32 + 16 + hi * 8]);
            v16h av3 = make_a16(&Vt[cur][48 + ln][p2 * 32 + hi * 8], &Vt[cur][48 + ln][p2 * 32 + 16 + hi * 8]);
            acc[0] = __builtin_amdgcn_wmma_f32_16x16x32_f16(false, av0, false, bp, (short)0, acc[0], false, false);
            acc[1] = __builtin_amdgcn_wmma_f32_16x16x32_f16(false, av1, false, bp, (short)0, acc[1], false, true);
            acc[2] = __builtin_amdgcn_wmma_f32_16x16x32_f16(false, av2, false, bp, (short)0, acc[2], false, true);
            acc[3] = __builtin_amdgcn_wmma_f32_16x16x32_f16(false, av3, false, bp, (short)0, acc[3], false, true);
        }

        // ---- convert + store the prefetched block into the other buffer ----
        if (have_next) {
#pragma unroll
            for (int j = 0; j < 4; ++j) {
                const int e = tid * 4 + j * 1024;
                const int kr = e >> 6, dc = e & 63;
#pragma unroll
                for (int t = 0; t < 4; ++t) {
                    const _Float16 h = (_Float16)g[j][t];
                    Ks[nxt][kr][dc + t] = h;
                    Vt[nxt][dc + t][kr] = h;
                }
            }
        }
        __syncthreads();                                 // single barrier per block
    }

    // ---- epilogue: O[q][d] = accᵀ / (0.9 * l)  (dropout rescale folded) ----
    const float inv_l = 1.0f / (0.9f * l_run);
    float* orow = optr + (size_t)(wave * 16 + ln) * D_DIM;
#pragma unroll
    for (int n = 0; n < 4; ++n) {
        v4f o0, o1;
#pragma unroll
        for (int r = 0; r < 4; ++r) { o0[r] = acc[n][r] * inv_l; o1[r] = acc[n][r + 4] * inv_l; }
        float* dst = orow + n * 16 + hi * 8;
        *(v4f*)dst       = o0;
        *(v4f*)(dst + 4) = o1;
    }
}

extern "C" void kernel_launch(void* const* d_in, const int* in_sizes, int n_in,
                              void* d_out, int out_size, void* d_ws, size_t ws_size,
                              hipStream_t stream) {
    (void)in_sizes; (void)n_in; (void)d_ws; (void)ws_size; (void)out_size;
    const float* x1 = (const float*)d_in[0];   // queries       [B,H,S,D] f32
    const float* x2 = (const float*)d_in[1];   // keys==values  [B,H,S,D] f32
    float* out = (float*)d_out;                // output        [B,H,S,D] f32

    dim3 grid(S_DIM / QTILE, B_DIM * H_DIM);   // (16, 32)
    dim3 block(256);
    fa_wmma_kernel<<<grid, block, 0, stream>>>(x1, x2, out);
}
